// AttentionPairBias_60516089201092
// MI455X (gfx1250) — compile-verified, matrix operand order checked by reading, not run on hardware
//
#include <hip/hip_runtime.h>
#include <hip/hip_bf16.h>

typedef __attribute__((ext_vector_type(16))) _Float16 v16h;
typedef __attribute__((ext_vector_type(8)))  float    v8f;

#define NSEQ 1024
#define CS   384
#define CZ   128
#define NH   16
#define DH   24
#define NN   (1024 * 1024)

__device__ __forceinline__ float sigmoidf_(float x) {
  return __builtin_amdgcn_rcpf(1.0f + __expf(-x));  // v_rcp_f32, avoids div_scale chain
}

// K index held by (element idx, lane-half hi) for 16-bit A/B operands of
// v_wmma_*_16x16x32 (ISA 7.12.2): VGPR r = idx>>1, sub-half c = idx&1.
__device__ __forceinline__ int wmma_kmap(int idx, int hi) {
  int r = idx >> 1, c = idx & 1;
  return (r < 4) ? (hi * 8 + 2 * r + c) : (16 + hi * 8 + 2 * (r - 4) + c);
}

// A tile: 16x32 (MxK), row-major source; per lane this is two contiguous
// 16-byte chunks -> compiler emits 2x global_load_b128.
__device__ __forceinline__ v16h load_a16(const _Float16* base, int ld, int row0, int k0) {
  int lane = threadIdx.x & 31;
  int m = lane & 15, hi = lane >> 4;
  v16h a;
#pragma unroll
  for (int idx = 0; idx < 16; ++idx)
    a[idx] = base[(size_t)(row0 + m) * ld + (k0 + wmma_kmap(idx, hi))];
  return a;
}

// Pre-swizzled B tile: tile tIdx holds 512 halves, lane-major, 16 contiguous
// halves per lane (32B aligned) -> one vector load.
__device__ __forceinline__ v16h load_bswz(const _Float16* swz, int tIdx) {
  int lane = threadIdx.x & 31;
  return *(const v16h*)(swz + (size_t)tIdx * 512 + lane * 16);
}

__device__ __forceinline__ v8f wmma_f16(v16h a, v16h b, v8f c) {
  return __builtin_amdgcn_wmma_f32_16x16x32_f16(false, a, false, b, (short)0, c, false, false);
}

// ---------------------------------------------------------------- B-operand swizzles
// f32 weight [K x N] row-major -> swizzled f16 B tiles, tile index nt*(K/32)+kt.
__global__ void swizzleB_kernel(const float* __restrict__ W, _Float16* __restrict__ dst,
                                int K, int N) {
  int tid = blockIdx.x * blockDim.x + threadIdx.x;
  if (tid >= K * N) return;
  int idx = tid & 15;
  int lane = (tid >> 4) & 31;
  int tile = tid >> 9;
  int ktiles = K >> 5;
  int kt = tile % ktiles;
  int nt = tile / ktiles;
  int hi = lane >> 4, n = lane & 15;
  int k = kt * 32 + wmma_kmap(idx, hi);
  dst[tid] = (_Float16)W[(size_t)k * N + nt * 16 + n];
}

// k16 [N, H*D] -> K^T B tiles per (head, 16-wide j tile): B[d(pad32)][j]
__global__ void swizzleK_kernel(const _Float16* __restrict__ k16, _Float16* __restrict__ dst) {
  int tid = blockIdx.x * blockDim.x + threadIdx.x;  // 16*64*512
  int idx = tid & 15;
  int lane = (tid >> 4) & 31;
  int tile = tid >> 9;        // 0..1023
  int h = tile >> 6, jt = tile & 63;
  int hi = lane >> 4, n = lane & 15;
  int d = wmma_kmap(idx, hi);
  dst[tid] = (d < DH) ? k16[(size_t)(jt * 16 + n) * CS + h * DH + d] : (_Float16)0.0f;
}

// v16 [N, H*D] -> V B tiles per (head, 32-wide j chunk, d half t): B[j][d]
__global__ void swizzleV_kernel(const _Float16* __restrict__ v16, _Float16* __restrict__ dst) {
  int tid = blockIdx.x * blockDim.x + threadIdx.x;  // 16*32*2*512
  int idx = tid & 15;
  int lane = (tid >> 4) & 31;
  int tile = tid >> 9;        // 0..1023
  int t = tile & 1, jc = (tile >> 1) & 31, h = tile >> 6;
  int hi = lane >> 4, n = lane & 15;
  int kk = wmma_kmap(idx, hi);
  int d = t * 16 + n;
  dst[tid] = (d < DH) ? v16[(size_t)(jc * 32 + kk) * CS + h * DH + d] : (_Float16)0.0f;
}

// ---------------------------------------------------------------- LayerNorm over 384
__global__ void ln384_kernel(const float* __restrict__ x, const float* __restrict__ gamma,
                             float* __restrict__ out32, _Float16* __restrict__ out16,
                             _Float16* __restrict__ raw16) {
  __shared__ float r1[512];
  __shared__ float r2[512];
  int row = blockIdx.x, t = threadIdx.x;
  float v = x[(size_t)row * CS + t];
  if (raw16) raw16[(size_t)row * CS + t] = (_Float16)v;
  r1[t] = v;
  r2[t] = v * v;
  if (t < 128) { r1[384 + t] = 0.f; r2[384 + t] = 0.f; }
  __syncthreads();
  for (int s = 256; s > 0; s >>= 1) {
    if (t < s) { r1[t] += r1[t + s]; r2[t] += r2[t + s]; }
    __syncthreads();
  }
  float mean = r1[0] * (1.0f / CS);
  float var  = r2[0] * (1.0f / CS) - mean * mean;
  float rstd = rsqrtf(var + 1e-5f);
  float ln = (v - mean) * rstd;
  if (out32) out32[(size_t)row * CS + t] = ln;
  if (out16) out16[(size_t)row * CS + t] = (_Float16)(ln * (gamma ? gamma[t] : 1.0f));
}

// ---------------------------------------------------------------- AdaLN fused dual-GEMM
// a = sigmoid(sln@Ws + bs) * aln + sln@Wns   -> f16
__global__ void adaln_kernel(const _Float16* __restrict__ sln, const _Float16* __restrict__ Ws,
                             const _Float16* __restrict__ Wns, const float* __restrict__ bs,
                             const float* __restrict__ aln, _Float16* __restrict__ aout) {
  int w = threadIdx.x >> 5;
  int tile = blockIdx.x * 8 + w;
  int i0 = (tile / (CS / 16)) * 16;
  int nt = tile % (CS / 16);
  int n0 = nt * 16;
  v8f c1 = {0, 0, 0, 0, 0, 0, 0, 0};
  v8f c2 = {0, 0, 0, 0, 0, 0, 0, 0};
  for (int kk = 0; kk < CS / 32; ++kk) {
    v16h a = load_a16(sln, CS, i0, kk * 32);
    c1 = wmma_f16(a, load_bswz(Ws, nt * (CS / 32) + kk), c1);
    c2 = wmma_f16(a, load_bswz(Wns, nt * (CS / 32) + kk), c2);
  }
  int lane = threadIdx.x & 31;
  int n = lane & 15, hi = lane >> 4;
  float bv = bs[n0 + n];
#pragma unroll
  for (int r = 0; r < 8; ++r) {
    int row = i0 + r + 8 * hi;
    float val = sigmoidf_(c1[r] + bv) * aln[(size_t)row * CS + n0 + n] + c2[r];
    aout[(size_t)row * CS + n0 + n] = (_Float16)val;
  }
}

// ---------------------------------------------------------------- generic 1024x384 @ 384x384
// mode 0: plain, mode 2: sigmoid. (acc+bias)*scale applied first.
__global__ void gemm384_kernel(const _Float16* __restrict__ A, const _Float16* __restrict__ Wswz,
                               const float* __restrict__ bias, float scale, int mode,
                               _Float16* __restrict__ out16, float* __restrict__ out32) {
  int w = threadIdx.x >> 5;
  int tile = blockIdx.x * 8 + w;
  int i0 = (tile / (CS / 16)) * 16;
  int nt = tile % (CS / 16);
  int n0 = nt * 16;
  v8f acc = {0, 0, 0, 0, 0, 0, 0, 0};
  for (int kk = 0; kk < CS / 32; ++kk)
    acc = wmma_f16(load_a16(A, CS, i0, kk * 32), load_bswz(Wswz, nt * (CS / 32) + kk), acc);
  int lane = threadIdx.x & 31;
  int n = lane & 15, hi = lane >> 4;
  float bv = bias ? bias[n0 + n] : 0.0f;
#pragma unroll
  for (int r = 0; r < 8; ++r) {
    int row = i0 + r + 8 * hi;
    float v = (acc[r] + bv) * scale;
    if (mode == 2) v = sigmoidf_(v);
    if (out16) out16[(size_t)row * CS + n0 + n] = (_Float16)v;
    if (out32) out32[(size_t)row * CS + n0 + n] = v;
  }
}

// ---------------------------------------------------------------- pair bias
// per flattened row R = i*1024+j: LN over 128 ch, *lnw+lnb, then [16x128] @ wb[128x16]
__global__ void pair_bias_kernel(const float* __restrict__ z, const float* __restrict__ lnw,
                                 const float* __restrict__ lnb, const _Float16* __restrict__ wbswz,
                                 float* __restrict__ biasout) {
  __shared__ _Float16 zs[8 * 16 * CZ];  // 32 KB, 4 KB per wave
  int lane = threadIdx.x & 31;
  int w = threadIdx.x >> 5;
  _Float16* zrow = zs + w * (16 * CZ);
  long tile = (long)blockIdx.x * 8 + w;
  long R0 = tile * 16;
  int i  = (int)(R0 >> 10);
  int j0 = (int)(R0 & 1023);
  float lw[4], lb[4];
#pragma unroll
  for (int t = 0; t < 4; ++t) { lw[t] = lnw[lane * 4 + t]; lb[t] = lnb[lane * 4 + t]; }
  v16h b0 = load_bswz(wbswz, 0);
  v16h b1 = load_bswz(wbswz, 1);
  v16h b2 = load_bswz(wbswz, 2);
  v16h b3 = load_bswz(wbswz, 3);
  for (int m = 0; m < 16; ++m) {
    const float* row = z + (size_t)(R0 + m) * CZ;
    float x[4], s = 0.f, sq = 0.f;
#pragma unroll
    for (int t = 0; t < 4; ++t) { x[t] = row[lane * 4 + t]; s += x[t]; sq += x[t] * x[t]; }
#pragma unroll
    for (int off = 1; off < 32; off <<= 1) {
      s += __shfl_xor(s, off, 32);
      sq += __shfl_xor(sq, off, 32);
    }
    float mean = s * (1.0f / CZ);
    float var  = sq * (1.0f / CZ) - mean * mean;
    float rstd = rsqrtf(var + 1e-5f);
#pragma unroll
    for (int t = 0; t < 4; ++t)
      zrow[m * CZ + lane * 4 + t] = (_Float16)(((x[t] - mean) * rstd) * lw[t] + lb[t]);
  }
  v8f acc = {0, 0, 0, 0, 0, 0, 0, 0};
  acc = wmma_f16(load_a16(zrow, CZ, 0, 0),  b0, acc);
  acc = wmma_f16(load_a16(zrow, CZ, 0, 32), b1, acc);
  acc = wmma_f16(load_a16(zrow, CZ, 0, 64), b2, acc);
  acc = wmma_f16(load_a16(zrow, CZ, 0, 96), b3, acc);
  int h = lane & 15, hi = lane >> 4;
  size_t base = (size_t)h * NN + (size_t)i * 1024 + j0;
#pragma unroll
  for (int r = 0; r < 8; ++r) biasout[base + r + 8 * hi] = acc[r];
}

// ---------------------------------------------------------------- flash attention
// one wave per (16-row i-tile, head); q pre-scaled by 1/sqrt(D); K/V pre-swizzled.
__global__ void flash_kernel(const _Float16* __restrict__ q, const _Float16* __restrict__ kswz,
                             const _Float16* __restrict__ vswz, const float* __restrict__ bias,
                             const float* __restrict__ g, _Float16* __restrict__ o16) {
  __shared__ _Float16 p_lds[16 * 32];  // 1 KB, single wave per block
  int lane = threadIdx.x;
  int i0 = blockIdx.x * 16;
  int h  = blockIdx.y;
  int m = lane & 15, hi = lane >> 4, n = lane & 15;
  v16h qa;
#pragma unroll
  for (int idx = 0; idx < 16; ++idx) {
    int d = wmma_kmap(idx, hi);
    qa[idx] = (d < DH) ? q[(size_t)(i0 + m) * CS + h * DH + d] : (_Float16)0.0f;
  }
  v8f Oa = {0, 0, 0, 0, 0, 0, 0, 0};
  v8f Ob = {0, 0, 0, 0, 0, 0, 0, 0};
  float rm[8], rs[8];
#pragma unroll
  for (int r = 0; r < 8; ++r) { rm[r] = -1e30f; rs[r] = 0.0f; }
  const float* brow = bias + (size_t)h * NN;

  for (int j0 = 0; j0 < NSEQ; j0 += 32) {
    int jt = j0 >> 4, jc = j0 >> 5;
    v16h kb0 = load_bswz(kswz, h * 64 + jt);
    v16h kb1 = load_bswz(kswz, h * 64 + jt + 1);
    v8f s0 = {0, 0, 0, 0, 0, 0, 0, 0};
    v8f s1 = {0, 0, 0, 0, 0, 0, 0, 0};
    s0 = wmma_f16(qa, kb0, s0);
    s1 = wmma_f16(qa, kb1, s1);
#pragma unroll
    for (int r = 0; r < 8; ++r) {
      int i = i0 + r + 8 * hi;
      s0[r] += brow[(size_t)i * 1024 + j0 + n];
      s1[r] += brow[(size_t)i * 1024 + j0 + 16 + n];
    }
    // online softmax: rows live across 16-lane halves (xor 1,2,4,8)
#pragma unroll
    for (int r = 0; r < 8; ++r) {
      float mx = fmaxf(s0[r], s1[r]);
      mx = fmaxf(mx, __shfl_xor(mx, 1, 32));
      mx = fmaxf(mx, __shfl_xor(mx, 2, 32));
      mx = fmaxf(mx, __shfl_xor(mx, 4, 32));
      mx = fmaxf(mx, __shfl_xor(mx, 8, 32));
      float mnew = fmaxf(rm[r], mx);
      float corr = __expf(rm[r] - mnew);
      float p0 = __expf(s0[r] - mnew);
      float p1 = __expf(s1[r] - mnew);
      float ps = p0 + p1;
      ps += __shfl_xor(ps, 1, 32);
      ps += __shfl_xor(ps, 2, 32);
      ps += __shfl_xor(ps, 4, 32);
      ps += __shfl_xor(ps, 8, 32);
      rs[r] = rs[r] * corr + ps;
      rm[r] = mnew;
      Oa[r] *= corr;
      Ob[r] *= corr;
      s0[r] = p0;
      s1[r] = p1;
    }
    // C-layout P -> LDS -> A-layout (wave-private; LDS in-order per wave)
#pragma unroll
    for (int r = 0; r < 8; ++r) {
      int row = r + 8 * hi;
      p_lds[row * 32 + n]      = (_Float16)s0[r];
      p_lds[row * 32 + 16 + n] = (_Float16)s1[r];
    }
    v16h pa;
#pragma unroll
    for (int idx = 0; idx < 16; ++idx)
      pa[idx] = p_lds[m * 32 + wmma_kmap(idx, hi)];
    v16h vb0 = load_bswz(vswz, (h * 32 + jc) * 2 + 0);
    v16h vb1 = load_bswz(vswz, (h * 32 + jc) * 2 + 1);
    Oa = wmma_f16(pa, vb0, Oa);
    Ob = wmma_f16(pa, vb1, Ob);
  }
#pragma unroll
  for (int r = 0; r < 8; ++r) {
    int i = i0 + r + 8 * hi;
    float inv = __builtin_amdgcn_rcpf(rs[r]);
    int d0 = n;
    float o0 = Oa[r] * inv * g[(size_t)i * CS + h * DH + d0];
    o16[(size_t)i * CS + h * DH + d0] = (_Float16)o0;
    int d1 = 16 + n;
    if (d1 < DH) {
      float o1 = Ob[r] * inv * g[(size_t)i * CS + h * DH + d1];
      o16[(size_t)i * CS + h * DH + d1] = (_Float16)o1;
    }
  }
}

// ---------------------------------------------------------------- final fused dual-GEMM
// out = sigmoid(sraw@Wsf + bs) * (o@Wo)   -> f32
__global__ void final_kernel(const _Float16* __restrict__ o16, const _Float16* __restrict__ Wo,
                             const _Float16* __restrict__ sraw, const _Float16* __restrict__ Wsf,
                             const float* __restrict__ bs, float* __restrict__ out) {
  int w = threadIdx.x >> 5;
  int tile = blockIdx.x * 8 + w;
  int i0 = (tile / (CS / 16)) * 16;
  int nt = tile % (CS / 16);
  int n0 = nt * 16;
  v8f c1 = {0, 0, 0, 0, 0, 0, 0, 0};
  v8f c2 = {0, 0, 0, 0, 0, 0, 0, 0};
  for (int kk = 0; kk < CS / 32; ++kk) {
    c1 = wmma_f16(load_a16(sraw, CS, i0, kk * 32), load_bswz(Wsf, nt * (CS / 32) + kk), c1);
    c2 = wmma_f16(load_a16(o16, CS, i0, kk * 32), load_bswz(Wo, nt * (CS / 32) + kk), c2);
  }
  int lane = threadIdx.x & 31;
  int n = lane & 15, hi = lane >> 4;
  float bv = bs[n0 + n];
#pragma unroll
  for (int r = 0; r < 8; ++r) {
    int row = i0 + r + 8 * hi;
    out[(size_t)row * CS + n0 + n] = sigmoidf_(c1[r] + bv) * c2[r];
  }
}

extern "C" void kernel_launch(void* const* d_in, const int* in_sizes, int n_in,
                              void* d_out, int out_size, void* d_ws, size_t ws_size,
                              hipStream_t stream) {
  (void)in_sizes; (void)n_in; (void)out_size; (void)ws_size;
  const float* a_i    = (const float*)d_in[0];
  const float* s_i    = (const float*)d_in[1];
  const float* z_ij   = (const float*)d_in[2];
  const float* lns_w  = (const float*)d_in[3];
  const float* ada_ws = (const float*)d_in[4];
  const float* ada_bs = (const float*)d_in[5];
  const float* ada_wns= (const float*)d_in[6];
  const float* wq     = (const float*)d_in[7];
  const float* bq     = (const float*)d_in[8];
  const float* wk     = (const float*)d_in[9];
  const float* wv     = (const float*)d_in[10];
  const float* lnb_w  = (const float*)d_in[11];
  const float* lnb_b  = (const float*)d_in[12];
  const float* wb     = (const float*)d_in[13];
  const float* wg     = (const float*)d_in[14];
  const float* wo     = (const float*)d_in[15];
  const float* wsf    = (const float*)d_in[16];
  const float* bsf    = (const float*)d_in[17];

  char* p = (char*)d_ws;
  auto alloc = [&](size_t bytes) -> char* {
    char* r = p;
    p += (bytes + 255) & ~(size_t)255;
    return r;
  };
  float*     bias32 = (float*)alloc((size_t)NH * NN * 4);        // 64 MB
  float*     aln32  = (float*)alloc((size_t)NSEQ * CS * 4);
  float*     g32    = (float*)alloc((size_t)NSEQ * CS * 4);
  _Float16*  sln16  = (_Float16*)alloc((size_t)NSEQ * CS * 2);
  _Float16*  sraw16 = (_Float16*)alloc((size_t)NSEQ * CS * 2);
  _Float16*  a16    = (_Float16*)alloc((size_t)NSEQ * CS * 2);
  _Float16*  q16    = (_Float16*)alloc((size_t)NSEQ * CS * 2);
  _Float16*  k16    = (_Float16*)alloc((size_t)NSEQ * CS * 2);
  _Float16*  v16    = (_Float16*)alloc((size_t)NSEQ * CS * 2);
  _Float16*  o16    = (_Float16*)alloc((size_t)NSEQ * CS * 2);
  _Float16*  wadaS  = (_Float16*)alloc((size_t)CS * CS * 2);   // swizzled B layouts
  _Float16*  wadnS  = (_Float16*)alloc((size_t)CS * CS * 2);
  _Float16*  wqS    = (_Float16*)alloc((size_t)CS * CS * 2);
  _Float16*  wkS    = (_Float16*)alloc((size_t)CS * CS * 2);
  _Float16*  wvS    = (_Float16*)alloc((size_t)CS * CS * 2);
  _Float16*  wgS    = (_Float16*)alloc((size_t)CS * CS * 2);
  _Float16*  woS    = (_Float16*)alloc((size_t)CS * CS * 2);
  _Float16*  wsfS   = (_Float16*)alloc((size_t)CS * CS * 2);
  _Float16*  wbS    = (_Float16*)alloc((size_t)CZ * NH * 2);
  _Float16*  kswz   = (_Float16*)alloc((size_t)NH * 64 * 512 * 2);      // 1 MB
  _Float16*  vswz   = (_Float16*)alloc((size_t)NH * 32 * 2 * 512 * 2);  // 1 MB

  const int WN = CS * CS;
  const int CB = (WN + 255) / 256;
  swizzleB_kernel<<<CB, 256, 0, stream>>>(ada_ws, wadaS, CS, CS);
  swizzleB_kernel<<<CB, 256, 0, stream>>>(ada_wns, wadnS, CS, CS);
  swizzleB_kernel<<<CB, 256, 0, stream>>>(wq, wqS, CS, CS);
  swizzleB_kernel<<<CB, 256, 0, stream>>>(wk, wkS, CS, CS);
  swizzleB_kernel<<<CB, 256, 0, stream>>>(wv, wvS, CS, CS);
  swizzleB_kernel<<<CB, 256, 0, stream>>>(wg, wgS, CS, CS);
  swizzleB_kernel<<<CB, 256, 0, stream>>>(wo, woS, CS, CS);
  swizzleB_kernel<<<CB, 256, 0, stream>>>(wsf, wsfS, CS, CS);
  swizzleB_kernel<<<(CZ * NH + 255) / 256, 256, 0, stream>>>(wb, wbS, CZ, NH);

  ln384_kernel<<<NSEQ, CS, 0, stream>>>(a_i, nullptr, aln32, nullptr, nullptr);
  ln384_kernel<<<NSEQ, CS, 0, stream>>>(s_i, lns_w, nullptr, sln16, sraw16);

  adaln_kernel<<<192, 256, 0, stream>>>(sln16, wadaS, wadnS, ada_bs, aln32, a16);

  // q gets bias + 1/sqrt(24) baked in; g gets sigmoid (f32 gate)
  gemm384_kernel<<<192, 256, 0, stream>>>(a16, wqS, bq, 0.20412414523193154f, 0, q16, nullptr);
  gemm384_kernel<<<192, 256, 0, stream>>>(a16, wkS, nullptr, 1.0f, 0, k16, nullptr);
  gemm384_kernel<<<192, 256, 0, stream>>>(a16, wvS, nullptr, 1.0f, 0, v16, nullptr);
  gemm384_kernel<<<192, 256, 0, stream>>>(a16, wgS, nullptr, 1.0f, 2, nullptr, g32);

  // re-swizzle K/V into B-operand tile layout (vector loads in flash loop)
  swizzleK_kernel<<<(NH * 64 * 512) / 256, 256, 0, stream>>>(k16, kswz);
  swizzleV_kernel<<<(NH * 64 * 512) / 256, 256, 0, stream>>>(v16, vswz);

  // 1M rows / (8 waves * 16 rows) = 8192 blocks; streams z_ij exactly once
  pair_bias_kernel<<<8192, 256, 0, stream>>>(z_ij, lnb_w, lnb_b, wbS, bias32);

  flash_kernel<<<dim3(NSEQ / 16, NH), 32, 0, stream>>>(q16, kswz, vswz, bias32, g32, o16);

  final_kernel<<<192, 256, 0, stream>>>(o16, woS, sraw16, wsfS, bsf, (float*)d_out);
}
